// AttentionInteractionBlockVN_45646912422492
// MI455X (gfx1250) — matrix operand order repaired and checked by reading, not compiled, not run on hardware
//
#include <hip/hip_runtime.h>
#include <hip/hip_bf16.h>
#include <math.h>

// ---------------------------------------------------------------------------
// AttentionInteractionBlockVN for MI455X (gfx1250, wave32, WMMA).
// - every dense linear -> bf16 WMMA GEMM (f32 accumulate)
// - edge vector channel is rank-1 (edge_vec_feat = ee_W (x) nv): edge_gvl /
//   vn_leaky_relu / evn collapse to per-(e,o) coefficient GEMMs + scalar math
// - all (M,64,3) vector tensors stored d-major (M,3,64) so every vn_linear
//   d-slice is a contiguous-row GEMM (vector loads); transposed only at the
//   x_vec input and out_vec output boundaries
// - GEMM: block = 8 waves sharing one 64-col weight strip staged in LDS as
//   plain [col][k] bf16; wave = 16x64 strip = 4 accumulators; A fragments are
//   one contiguous 16-float run (4x b128), software-pipelined one K-step ahead
// ---------------------------------------------------------------------------

#define HSD 256
#define SLOPE_F 0.01f
#define EPS_F 1e-6f
#define LNEPS_F 1e-5f
#define VECEPS_F 1e-7f
#define CUTOFF_F 10.0f
#define NGAUSS 60
#define PI_F 3.14159265358979f
#define MAXK 320

typedef __attribute__((ext_vector_type(16))) __bf16 v16bf;
typedef __attribute__((ext_vector_type(8)))  __bf16 v8bf;
typedef __attribute__((ext_vector_type(8)))  float  v8f;

__device__ __forceinline__ float sigmoidf_(float x){ return 1.0f/(1.0f+__expf(-x)); }
__device__ __forceinline__ float leakyf_(float x){ return x < 0.0f ? SLOPE_F*x : x; }

__device__ __forceinline__ void pack8a(v16bf& dst, int base, float4 a, float4 b){
  dst[base+0]=(__bf16)a.x; dst[base+1]=(__bf16)a.y;
  dst[base+2]=(__bf16)a.z; dst[base+3]=(__bf16)a.w;
  dst[base+4]=(__bf16)b.x; dst[base+5]=(__bf16)b.y;
  dst[base+6]=(__bf16)b.z; dst[base+7]=(__bf16)b.w;
}

// C[m,n] = sum_k A[m,k]*W[n,k] (+bias[n]).  W row-major (Nout,K).
// A rows contiguous (stride aRow), optional row gather.  C rows stride cRow.
// M%16==0, K%32==0, Nout%64==0 at every call site.
// grid = (Nout/64, ceil(M/128)); block = 256 (8 waves, one 16x64 strip each).
__global__ __launch_bounds__(256) void gemm_wmma_bf16(
    const float* __restrict__ A, long aRow, const int* __restrict__ gather,
    const float* __restrict__ B,
    const float* __restrict__ bias,
    float* __restrict__ C, long cRow,
    int M, int K)
{
  __shared__ __bf16 ldsB[64 * MAXK];              // strip weights, bf16, [col][k]
  const int ntb = blockIdx.x << 6;                // strip start column

  // ---- cooperative fill: 64 x K strip -> bf16 LDS (coalesced, vectorized)
  for (int t = threadIdx.x * 8; t < 64 * K; t += 256 * 8) {
    int c = t / K, k = t - c * K;                 // K%8==0 -> run stays in-row
    const float4* g = (const float4*)(B + (long)(ntb + c) * K + k);
    float4 f0 = g[0], f1 = g[1];
    v8bf pk;
    pk[0]=(__bf16)f0.x; pk[1]=(__bf16)f0.y; pk[2]=(__bf16)f0.z; pk[3]=(__bf16)f0.w;
    pk[4]=(__bf16)f1.x; pk[5]=(__bf16)f1.y; pk[6]=(__bf16)f1.z; pk[7]=(__bf16)f1.w;
    *(v8bf*)&ldsB[c * K + k] = pk;
  }
  __syncthreads();

  const int mt = blockIdx.y * 8 + (threadIdx.x >> 5);
  if (mt * 16 >= M) return;
  const int lane = threadIdx.x & 31;
  const int half = lane >> 4;                     // 0: lanes 0-15, 1: 16-31
  const int l    = lane & 15;

  int row = mt * 16 + l;
  if (gather) row = gather[row];
  const float* apk = A + (long)row * aRow + half * 16;

  v8f acc[4];
#pragma unroll
  for (int j = 0; j < 4; ++j) acc[j] = (v8f){0.f,0.f,0.f,0.f,0.f,0.f,0.f,0.f};

  // prime A pipeline: one contiguous 16-float run (k = half*16..half*16+15)
  float4 a0 = *(const float4*)(apk +  0), a1 = *(const float4*)(apk +  4);
  float4 a2 = *(const float4*)(apk +  8), a3 = *(const float4*)(apk + 12);

  for (int kk = 0; kk < K; kk += 32) {
    float4 n0 = a0, n1 = a1, n2 = a2, n3 = a3;
    if (kk + 32 < K) {                            // prefetch next K-step
      const float* p = apk + kk + 32;
      n0 = *(const float4*)(p +  0); n1 = *(const float4*)(p +  4);
      n2 = *(const float4*)(p +  8); n3 = *(const float4*)(p + 12);
    }
    v16bf af;
    pack8a(af, 0, a0, a1);
    pack8a(af, 8, a2, a3);
#pragma unroll
    for (int j = 0; j < 4; ++j) {
      // B fragment: 16 contiguous bf16 (32B) from LDS, col = j*16+l
      const v16bf bfv = *(const v16bf*)&ldsB[(j*16 + l) * K + kk + 16*half];
      acc[j] = __builtin_amdgcn_wmma_f32_16x16x32_bf16(
          false, af, false, bfv, (short)0, acc[j], false, false);
    }
    a0 = n0; a1 = n1; a2 = n2; a3 = n3;
  }

#pragma unroll
  for (int j = 0; j < 4; ++j) {
    int ncol = ntb + j*16 + l;
    float badd = bias ? bias[ncol] : 0.0f;
#pragma unroll
    for (int r = 0; r < 8; ++r) {
      int m = mt * 16 + r + 8 * half;             // C tile: VGPR r -> M=r / r+8
      C[(long)m * cRow + ncol] = acc[j][r] + badd;
    }
  }
}

// ---------------------------------------------------------------------------
// Elementwise / small kernels.  Vector tensors are d-major: (M,3,64).
// ---------------------------------------------------------------------------
__global__ void kzero(float* __restrict__ p, long n){
  long i = blockIdx.x * (long)blockDim.x + threadIdx.x;
  if (i < n) p[i] = 0.0f;
}

// (N,64,3) -> (N,3,64)
__global__ void ktranspose_vec(const float* __restrict__ in, float* __restrict__ out, long n){
  long i = blockIdx.x * (long)blockDim.x + threadIdx.x;
  if (i >= n) return;
  long m = i / 192; int rem = (int)(i % 192);
  int d = rem >> 6, o = rem & 63;
  out[i] = in[m*192 + o*3 + d];
}

// w1e[o] = sum_i Wv1_edge[o,i]*ee_W[i] ;  q[o] = sum_i Wv2_edge[o,i]*w1e[i]
__global__ void kprep(const float* __restrict__ Wv1e, const float* __restrict__ Wv2e,
                      const float* __restrict__ eeW,
                      float* __restrict__ w1e, float* __restrict__ qv){
  __shared__ float s[64];
  int o = threadIdx.x;
  float acc = 0.f;
  for (int i = 0; i < 64; ++i) acc += Wv1e[o*64+i] * eeW[i];
  s[o] = acc; w1e[o] = acc;
  __syncthreads();
  float q = 0.f;
  for (int i = 0; i < 64; ++i) q += Wv2e[o*64+i] * s[i];
  qv[o] = q;
}

// Per edge: dist, smearing, rank-1 vnorm, normalized vec, cutoff. blockDim=64.
// ecat per edge (128): [ |w1e[o]|*||nv|| (64) | smear(60)+feat(4) ]
__global__ void kedge_feat(const float* __restrict__ ev, const float* __restrict__ efeat,
                           const float* __restrict__ w1e,
                           float* __restrict__ ecat, float* __restrict__ nvb,
                           float* __restrict__ ccut){
  long e = blockIdx.x; int t = threadIdx.x;
  float x = ev[e*3+0], y = ev[e*3+1], z = ev[e*3+2];
  float dist = sqrtf(x*x + y*y + z*z);
  float inv  = 1.0f / (dist + VECEPS_F);
  float nrm  = dist * inv;                   // ||nv||
  ecat[e*128 + t] = fabsf(w1e[t]) * nrm;
  const float delta = CUTOFF_F / (NGAUSS - 1);
  const float coeff = -0.5f / (delta * delta);
  if (t < NGAUSS) {
    float diff = dist - (float)t * delta;
    ecat[e*128 + 64 + t] = __expf(coeff * diff * diff);
  } else {
    ecat[e*128 + 64 + t] = efeat[e*4 + (t - NGAUSS)];
  }
  if (t < 3)  nvb[e*4 + t] = ev[e*3 + t] * inv;
  if (t == 3) nvb[e*4 + 3] = nrm * nrm;
  if (t == 0) {
    float C = 0.5f * (__cosf(dist * PI_F / CUTOFF_F) + 1.0f);
    ccut[e] = (dist <= CUTOFF_F) ? C : 0.0f;
  }
}

// cat[m,0:64] = ||v[m,:,o]|| (d-major vec), cat[m,64:320] = sca.  blockDim=320
__global__ void knorm_concat(const float* __restrict__ v, const float* __restrict__ sca,
                             float* __restrict__ cat){
  long m = blockIdx.x; int t = threadIdx.x;
  if (t < 64) {
    const float* p = v + m*192 + t;
    float a = p[0], b = p[64], c = p[128];
    cat[m*320 + t] = sqrtf(a*a + b*b + c*c);
  } else {
    cat[m*320 + t] = sca[m*256 + (t - 64)];
  }
}

// vout[m,d,o] = sigmoid(gl[m,o]) * vin[m,d,o]   (d-major in & out)
__global__ void kgate_mul(const float* __restrict__ gl, const float* __restrict__ vin,
                          float* __restrict__ vout, long n){
  long i = blockIdx.x * (long)blockDim.x + threadIdx.x;
  if (i >= n) return;
  long m = i / 192; int o = (int)(i % 192) & 63;
  vout[i] = vin[i] * sigmoidf_(gl[m*64 + o]);
}

// final gate: read d-major, write reference (N,64,3) layout
__global__ void kgate_out(const float* __restrict__ gl, const float* __restrict__ vin,
                          float* __restrict__ out, long n){
  long i = blockIdx.x * (long)blockDim.x + threadIdx.x;
  if (i >= n) return;
  long m = i / 192; int rem = (int)(i % 192);
  int d = rem >> 6, o = rem & 63;
  out[m*192 + o*3 + d] = vin[i] * sigmoidf_(gl[m*64 + o]);
}

__global__ void kleaky_inplace(float* __restrict__ x, long n){
  long i = blockIdx.x * (long)blockDim.x + threadIdx.x;
  if (i < n) x[i] = leakyf_(x[i]);
}

// a[e,o] = sigmoid(gate_e[e,o]) * q[o]
__global__ void k_acoef(const float* __restrict__ gate_e, const float* __restrict__ qv,
                        float* __restrict__ a, long n){
  long i = blockIdx.x * (long)blockDim.x + threadIdx.x;
  if (i < n) a[i] = sigmoidf_(gate_e[i]) * qv[i & 63];
}

// rank-1 vn_leaky_relu on coefficients (vectors are coef * nv[e])
__global__ void k_vnl_coef(float* __restrict__ a, const float* __restrict__ dcoef,
                           const float* __restrict__ nvb, long n){
  long i = blockIdx.x * (long)blockDim.x + threadIdx.x;
  if (i >= n) return;
  long e = i >> 6;
  float av = a[i], dv = dcoef[i], n2 = nvb[e*4 + 3];
  float dot = av * dv * n2;
  float dsq = dv * dv * n2;
  float aneg = av - (dot / (dsq + EPS_F)) * dv;
  float sel  = (dot >= 0.0f) ? av : aneg;
  a[i] = SLOPE_F * av + (1.0f - SLOPE_F) * sel;
}

// y_sca[e,c] = n_sca[col[e],c] * sca_lin[e,c]   (in place over sca_lin)
__global__ void ky_sca(float* __restrict__ sca_lin, const float* __restrict__ n_sca,
                       const int* __restrict__ col, long n){
  long i = blockIdx.x * (long)blockDim.x + threadIdx.x;
  if (i >= n) return;
  long e = i >> 8; int c = (int)(i & 255);
  sca_lin[i] *= n_sca[(long)col[e]*256 + c];
}

// y_vec[e,d,o] = e2n[e,o]*n_vec[col,d,o] + n2e[e,o]*r[e,o]*nv[e,d]  (d-major)
__global__ void ky_vec(float* __restrict__ yv, const float* __restrict__ e2n,
                       const float* __restrict__ n2e, const float* __restrict__ r,
                       const float* __restrict__ nvb, const float* __restrict__ n_vec,
                       const int* __restrict__ col, long n){
  long i = blockIdx.x * (long)blockDim.x + threadIdx.x;
  if (i >= n) return;
  long e = i / 192; int rem = (int)(i % 192);
  int d = rem >> 6, o = rem & 63;
  yv[i] = e2n[e*64+o] * n_vec[(long)col[e]*192 + rem]
        + n2e[e*64+o] * r[e*64+o] * nvb[e*4 + d];
}

__global__ void kscatter_sca(const float* __restrict__ ms, const float* __restrict__ ccut,
                             const int* __restrict__ row, float* __restrict__ aggr, long n){
  long i = blockIdx.x * (long)blockDim.x + threadIdx.x;
  if (i >= n) return;
  long e = i >> 8; int c = (int)(i & 255);
  atomicAdd(&aggr[(long)row[e]*256 + c], ms[i] * ccut[e]);
}

// msg_vec scatter, d-major aggregation buffer
__global__ void kscatter_vec(const float* __restrict__ gate_o, const float* __restrict__ v2,
                             const float* __restrict__ ccut, const int* __restrict__ row,
                             float* __restrict__ aggr, long n){
  long i = blockIdx.x * (long)blockDim.x + threadIdx.x;
  if (i >= n) return;
  long e = i / 192; int rem = (int)(i % 192); int o = rem & 63;
  float val = sigmoidf_(gate_o[e*64+o]) * v2[i] * ccut[e];
  atomicAdd(&aggr[(long)row[e]*192 + rem], val);
}

// layernorm over 256 scalars + leaky_relu (blockDim = 256)
__global__ void kln_sca(const float* __restrict__ a, const float* __restrict__ b,
                        const float* __restrict__ w, const float* __restrict__ bb,
                        float* __restrict__ out){
  __shared__ float ssum[256]; __shared__ float ssq[256];
  long n = blockIdx.x; int t = threadIdx.x;
  float x = a[n*256+t] + b[n*256+t];
  ssum[t] = x; ssq[t] = x*x;
  __syncthreads();
  for (int s = 128; s > 0; s >>= 1) {
    if (t < s) { ssum[t] += ssum[t+s]; ssq[t] += ssq[t+s]; }
    __syncthreads();
  }
  float mu  = ssum[0] * (1.0f/256.0f);
  float var = ssq[0]  * (1.0f/256.0f) - mu*mu;
  float y = (x - mu) * rsqrtf(var + LNEPS_F) * w[t] + bb[t];
  out[n*256+t] = leakyf_(y);
}

// layernorm over 192 vector comps (blockDim = 192); buffers d-major, w/b (64,3)
__global__ void kln_vec(const float* __restrict__ a, const float* __restrict__ b,
                        const float* __restrict__ w, const float* __restrict__ bb,
                        float* __restrict__ out){
  __shared__ float ssum[192]; __shared__ float ssq[192];
  __shared__ float stat[2];
  long n = blockIdx.x; int t = threadIdx.x;
  float x = a[n*192+t] + b[n*192+t];
  ssum[t] = x; ssq[t] = x*x;
  __syncthreads();
  for (int s = 96; s >= 3; s >>= 1) {
    if (t < s) { ssum[t] += ssum[t+s]; ssq[t] += ssq[t+s]; }
    __syncthreads();
  }
  if (t == 0) {
    float su = ssum[0] + ssum[1] + ssum[2];
    float sq = ssq[0]  + ssq[1]  + ssq[2];
    float mu = su * (1.0f/192.0f);
    stat[0] = mu;
    stat[1] = rsqrtf(sq * (1.0f/192.0f) - mu*mu + LNEPS_F);
  }
  __syncthreads();
  int d = t >> 6, o = t & 63;
  out[n*192+t] = (x - stat[0]) * stat[1] * w[o*3+d] + bb[o*3+d];
}

// full 3-D vn_leaky_relu (one thread per (n,o)); d-major buffers
__global__ void k_vnl_full(const float* __restrict__ x, const float* __restrict__ d,
                           float* __restrict__ out, long n64){
  long i = blockIdx.x * (long)blockDim.x + threadIdx.x;
  if (i >= n64) return;
  long n = i >> 6; int o = (int)(i & 63);
  long b = n*192 + o;
  float x0=x[b], x1=x[b+64], x2=x[b+128];
  float d0=d[b], d1=d[b+64], d2=d[b+128];
  float dot = x0*d0 + x1*d1 + x2*d2;
  float dsq = d0*d0 + d1*d1 + d2*d2;
  float f = dot / (dsq + EPS_F);
  bool pos = (dot >= 0.0f);
  float r0 = pos ? x0 : x0 - f*d0;
  float r1 = pos ? x1 : x1 - f*d1;
  float r2 = pos ? x2 : x2 - f*d2;
  out[b]     = SLOPE_F*x0 + (1.0f-SLOPE_F)*r0;
  out[b+64]  = SLOPE_F*x1 + (1.0f-SLOPE_F)*r1;
  out[b+128] = SLOPE_F*x2 + (1.0f-SLOPE_F)*r2;
}

// ---------------------------------------------------------------------------
// Host orchestration
// ---------------------------------------------------------------------------
extern "C" void kernel_launch(void* const* d_in, const int* in_sizes, int n_in,
                              void* d_out, int out_size, void* d_ws, size_t ws_size,
                              hipStream_t stream) {
  (void)n_in; (void)out_size; (void)ws_size;
  const int N = in_sizes[0] / HSD;       // 10000
  const int E = in_sizes[4] / 3;         // 160000

  // inputs (recursive insertion-order flatten of setup_inputs dict)
  const float* x_sca = (const float*)d_in[0];
  const float* x_vec = (const float*)d_in[1];
  const int*   eidx  = (const int*)  d_in[2];
  const float* efeat = (const float*)d_in[3];
  const float* evec  = (const float*)d_in[4];
  // d_in[5] = idx_protein (unused)
  const float* eeW   = (const float*)d_in[6];
  const float *Wv1n=(const float*)d_in[7],  *Wv2n=(const float*)d_in[8],
              *Wsn =(const float*)d_in[9],  *Wgn =(const float*)d_in[10], *bgn =(const float*)d_in[11];
  const float *Wv1e=(const float*)d_in[12], *Wv2e=(const float*)d_in[13],
              *Wse =(const float*)d_in[14], *Wge =(const float*)d_in[15], *bge =(const float*)d_in[16];
  const float *Wd  =(const float*)d_in[17];
  const float *scaW=(const float*)d_in[18], *scab=(const float*)d_in[19];
  const float *e2nW=(const float*)d_in[20], *e2nb=(const float*)d_in[21];
  const float *n2eW=(const float*)d_in[22], *n2eb=(const float*)d_in[23];
  const float *evnW=(const float*)d_in[24];
  const float *Wv1o=(const float*)d_in[25], *Wv2o=(const float*)d_in[26],
              *Wso =(const float*)d_in[27], *Wgo =(const float*)d_in[28], *bgo =(const float*)d_in[29];
  const float *Wv1c=(const float*)d_in[30], *Wv2c=(const float*)d_in[31],
              *Wsc =(const float*)d_in[32], *Wgc =(const float*)d_in[33], *bgc =(const float*)d_in[34];
  const float *aWd =(const float*)d_in[35];
  const float *Wv1f=(const float*)d_in[36], *Wv2f=(const float*)d_in[37],
              *Wsf =(const float*)d_in[38], *Wgf =(const float*)d_in[39], *bgf =(const float*)d_in[40];
  const float *lnsw=(const float*)d_in[41], *lnsb=(const float*)d_in[42];
  const float *lnvw=(const float*)d_in[43], *lnvb=(const float*)d_in[44];

  const int* rowIdx = eidx;        // edge_index[0]
  const int* colIdx = eidx + E;    // edge_index[1]

  float* ws = (float*)d_ws;
  size_t off = 0;
  auto alloc = [&](size_t n){ float* p = ws + off; off += n; return p; };

  // node_gvl (vector buffers d-major (M,3,64))
  float* xvT   = alloc((size_t)N*192);
  float* nv1   = alloc((size_t)N*192);
  float* ncat  = alloc((size_t)N*320);
  float* n_sca = alloc((size_t)N*256);
  float* n_gl  = alloc((size_t)N*64);
  float* n_vec = alloc((size_t)N*192);
  // centroid
  float* cnv1  = alloc((size_t)N*192);
  float* ccat  = alloc((size_t)N*320);
  float* c_sca = alloc((size_t)N*256);
  float* c_gl  = alloc((size_t)N*64);
  float* c_vec = alloc((size_t)N*192);
  // edge pipeline
  float* w1e   = alloc(64);
  float* qv    = alloc(64);
  float* ecat  = alloc((size_t)E*128);
  float* nvb   = alloc((size_t)E*4);
  float* ccut  = alloc((size_t)E);
  float* eslin = alloc((size_t)E*64);   // -> e_sca_act (in place)
  float* gatee = alloc((size_t)E*64);
  float* acoef = alloc((size_t)E*64);   // -> activated coef (in place)
  float* dcoef = alloc((size_t)E*64);
  float* rcoef = alloc((size_t)E*64);
  float* scal  = alloc((size_t)E*256);  // -> y_sca (in place)
  float* e2n   = alloc((size_t)E*64);
  float* n2e   = alloc((size_t)E*64);
  float* yvec  = alloc((size_t)E*192);
  float* vy    = alloc((size_t)E*192);
  float* ycat  = alloc((size_t)E*320);
  float* mslin = alloc((size_t)E*256);
  float* gateo = alloc((size_t)E*64);
  float* v2    = alloc((size_t)E*192);
  // aggregation + output stage
  float* aggs  = alloc((size_t)N*256);
  float* aggv  = alloc((size_t)N*192);
  float* lns   = alloc((size_t)N*256);
  float* lnv   = alloc((size_t)N*192);
  float* dbuf  = alloc((size_t)N*192);
  float* actv  = alloc((size_t)N*192);
  float* ov1   = alloc((size_t)N*192);
  float* ocat  = alloc((size_t)N*320);
  float* ogl   = alloc((size_t)N*64);
  float* ov2   = alloc((size_t)N*192);

  float* out_sca = (float*)d_out;            // (N,256)
  float* out_vec = out_sca + (size_t)N*256;  // (N,64,3), written by kgate_out

  auto gemm = [&](const float* A, long aRow, const int* gth,
                  const float* B, int K, const float* bias,
                  float* C, long cRow, int M, int Nout){
    dim3 grid(Nout / 64, (M/16 + 7) / 8);
    gemm_wmma_bf16<<<grid, 256, 0, stream>>>(A, aRow, gth, B, bias, C, cRow, M, K);
  };
  auto nb = [](long n){ return (int)((n + 255) / 256); };

  // ---------------- Phase A: node_gvl + centroid (gv_linear 256/64 -> 256/64)
  ktranspose_vec<<<nb((long)N*192), 256, 0, stream>>>(x_vec, xvT, (long)N*192);
  for (int d = 0; d < 3; ++d)
    gemm(xvT + d*64, 192, nullptr, Wv1n, 64, nullptr, nv1 + d*64, 192, N, 64);
  knorm_concat<<<N, 320, 0, stream>>>(nv1, x_sca, ncat);
  gemm(ncat, 320, nullptr, Wsn, 320, nullptr, n_sca, 256, N, 256);
  for (int d = 0; d < 3; ++d)
    gemm(nv1 + d*64, 192, nullptr, Wv2n, 64, nullptr, n_vec + d*64, 192, N, 64);
  gemm(n_sca, 256, nullptr, Wgn, 256, bgn, n_gl, 64, N, 64);
  kgate_mul<<<nb((long)N*192), 256, 0, stream>>>(n_gl, n_vec, n_vec, (long)N*192);

  for (int d = 0; d < 3; ++d)
    gemm(xvT + d*64, 192, nullptr, Wv1c, 64, nullptr, cnv1 + d*64, 192, N, 64);
  knorm_concat<<<N, 320, 0, stream>>>(cnv1, x_sca, ccat);
  gemm(ccat, 320, nullptr, Wsc, 320, nullptr, c_sca, 256, N, 256);
  for (int d = 0; d < 3; ++d)
    gemm(cnv1 + d*64, 192, nullptr, Wv2c, 64, nullptr, c_vec + d*64, 192, N, 64);
  gemm(c_sca, 256, nullptr, Wgc, 256, bgc, c_gl, 64, N, 64);
  kgate_mul<<<nb((long)N*192), 256, 0, stream>>>(c_gl, c_vec, c_vec, (long)N*192);

  // ---------------- Phase B: edge message module (rank-1 vector channel)
  kprep<<<1, 64, 0, stream>>>(Wv1e, Wv2e, eeW, w1e, qv);
  kedge_feat<<<E, 64, 0, stream>>>(evec, efeat, w1e, ecat, nvb, ccut);
  gemm(ecat, 128, nullptr, Wse, 128, nullptr, eslin, 64, E, 64);
  gemm(eslin, 64, nullptr, Wge, 64, bge, gatee, 64, E, 64);
  kleaky_inplace<<<nb((long)E*64), 256, 0, stream>>>(eslin, (long)E*64);      // e_sca act
  k_acoef<<<nb((long)E*64), 256, 0, stream>>>(gatee, qv, acoef, (long)E*64);  // gated e_vec coef
  gemm(acoef, 64, nullptr, Wd, 64, nullptr, dcoef, 64, E, 64);
  k_vnl_coef<<<nb((long)E*64), 256, 0, stream>>>(acoef, dcoef, nvb, (long)E*64);
  gemm(acoef, 64, nullptr, evnW, 64, nullptr, rcoef, 64, E, 64);              // evn coef
  gemm(eslin, 64, nullptr, scaW, 64, scab, scal, 256, E, 256);
  ky_sca<<<nb((long)E*256), 256, 0, stream>>>(scal, n_sca, colIdx, (long)E*256);
  gemm(eslin, 64, nullptr, e2nW, 64, e2nb, e2n, 64, E, 64);
  gemm(n_sca, 256, colIdx, n2eW, 256, n2eb, n2e, 64, E, 64);                  // gathered rows
  ky_vec<<<nb((long)E*192), 256, 0, stream>>>(yvec, e2n, n2e, rcoef, nvb, n_vec, colIdx, (long)E*192);
  // out_gvl on edges
  for (int d = 0; d < 3; ++d)
    gemm(yvec + d*64, 192, nullptr, Wv1o, 64, nullptr, vy + d*64, 192, E, 64);
  knorm_concat<<<E, 320, 0, stream>>>(vy, scal, ycat);
  gemm(ycat, 320, nullptr, Wso, 320, nullptr, mslin, 256, E, 256);
  gemm(mslin, 256, nullptr, Wgo, 256, bgo, gateo, 64, E, 64);
  for (int d = 0; d < 3; ++d)
    gemm(vy + d*64, 192, nullptr, Wv2o, 64, nullptr, v2 + d*64, 192, E, 64);
  // cutoff scaling + scatter-sum into node accumulators (L2-resident atomics)
  kzero<<<nb((long)N*256), 256, 0, stream>>>(aggs, (long)N*256);
  kzero<<<nb((long)N*192), 256, 0, stream>>>(aggv, (long)N*192);
  kscatter_sca<<<nb((long)E*256), 256, 0, stream>>>(mslin, ccut, rowIdx, aggs, (long)E*256);
  kscatter_vec<<<nb((long)E*192), 256, 0, stream>>>(gateo, v2, ccut, rowIdx, aggv, (long)E*192);

  // ---------------- Phase C: layernorms, activations, output gv_linear
  kln_sca<<<N, 256, 0, stream>>>(c_sca, aggs, lnsw, lnsb, lns);
  kln_vec<<<N, 192, 0, stream>>>(c_vec, aggv, lnvw, lnvb, lnv);
  for (int d = 0; d < 3; ++d)
    gemm(lnv + d*64, 192, nullptr, aWd, 64, nullptr, dbuf + d*64, 192, N, 64);
  k_vnl_full<<<nb((long)N*64), 256, 0, stream>>>(lnv, dbuf, actv, (long)N*64);
  for (int d = 0; d < 3; ++d)
    gemm(actv + d*64, 192, nullptr, Wv1f, 64, nullptr, ov1 + d*64, 192, N, 64);
  knorm_concat<<<N, 320, 0, stream>>>(ov1, lns, ocat);
  gemm(ocat, 320, nullptr, Wsf, 320, nullptr, out_sca, 256, N, 256);          // final scalar out
  gemm(out_sca, 256, nullptr, Wgf, 256, bgf, ogl, 64, N, 64);
  for (int d = 0; d < 3; ++d)
    gemm(ov1 + d*64, 192, nullptr, Wv2f, 64, nullptr, ov2 + d*64, 192, N, 64);
  kgate_out<<<nb((long)N*192), 256, 0, stream>>>(ogl, ov2, out_vec, (long)N*192); // final vec out
}
